// EvenLamerGCN_73504070303821
// MI455X (gfx1250) — matrix operand order repaired and checked by reading, not compiled, not run on hardware
//
#include <hip/hip_runtime.h>

// ---------------- types for WMMA ----------------
typedef __attribute__((ext_vector_type(16))) __bf16 v16bf;
typedef __attribute__((ext_vector_type(8)))  __bf16 v8bf;
typedef __attribute__((ext_vector_type(8)))  float  v8f;

#define DIN 512
#define DH  128
#define DOUT 40
#define DOUTP 48   // padded to 3x16 column tiles

// ---------------- small utility kernels ----------------

__global__ void k_fill0(float4* __restrict__ p, int n4) {
  int i = blockIdx.x * 256 + threadIdx.x;
  if (i < n4) p[i] = make_float4(0.f, 0.f, 0.f, 0.f);
}

// W1 [512][128] f32 -> W1t bf16 [128][512] (column-major weights, K contiguous)
__global__ void k_w1t(const float* __restrict__ W1, __bf16* __restrict__ W1t) {
  int i = blockIdx.x * 256 + threadIdx.x;           // 0 .. 128*512-1
  if (i >= DH * DIN) return;
  int n = i / DIN, k = i % DIN;
  W1t[i] = (__bf16)W1[k * DH + n];
}

// W2 [128][40] f32 -> W2t bf16 [48][128], zero-padded columns 40..47
__global__ void k_w2t(const float* __restrict__ W2, __bf16* __restrict__ W2t) {
  int i = blockIdx.x * 256 + threadIdx.x;           // 0 .. 48*128-1
  if (i >= DOUTP * DH) return;
  int n = i / DH, k = i % DH;
  W2t[i] = (__bf16)((n < DOUT) ? W2[k * DOUT + n] : 0.f);
}

__global__ void k_deg_init(float* __restrict__ deg, int N) {
  int i = blockIdx.x * 256 + threadIdx.x;
  if (i < N) deg[i] = 1.0f;                         // self-loop
}

__global__ void k_deg_edges(const int* __restrict__ dst, float* __restrict__ deg, int E) {
  int e = blockIdx.x * 256 + threadIdx.x;
  if (e < E) atomicAdd(&deg[dst[e]], 1.0f);
}

__global__ void k_dinv(const float* __restrict__ deg, float* __restrict__ dinv, int N) {
  int i = blockIdx.x * 256 + threadIdx.x;
  if (i < N) dinv[i] = rsqrtf(deg[i]);
}

// ---------------- GEMM1: H = X @ W1 (unscaled), bf16 WMMA ----------------
// grid: ceil(N/128) blocks of 256 threads (8 waves). Wave w owns column tile w
// (16 of 128 cols), computes all 8 row-subtiles of the 128-row block.
__global__ __launch_bounds__(256) void k_gemm1(const float* __restrict__ X,
                                               const __bf16* __restrict__ W1t,
                                               float* __restrict__ H, int N) {
  __shared__ __bf16 lA[128][48];                    // 32 K-cols staged, pitch 48 for alignment
  const int tid  = threadIdx.x;
  const int wave = tid >> 5;
  const int lane = tid & 31;
  const int n16  = lane & 15;
  const int half = lane >> 4;
  const int rowBase = blockIdx.x * 128;

  v8f z = {};
  v8f acc[8];
  for (int i = 0; i < 8; ++i) acc[i] = z;

  const __bf16* wrow = W1t + (size_t)(wave * 16 + n16) * DIN;

  for (int kb = 0; kb < DIN; kb += 32) {
    __syncthreads();
    // stage 128x32 f32 tile of X as bf16
    for (int it = 0; it < 4; ++it) {
      int i = tid + it * 256;                       // 0..1023 (float4 index)
      int r = i >> 3, c4 = i & 7;
      int gr = rowBase + r; if (gr >= N) gr = N - 1;
      float4 f = *(const float4*)(X + (size_t)gr * DIN + kb + c4 * 4);
      __bf16* p = &lA[r][c4 * 4];
      p[0] = (__bf16)f.x; p[1] = (__bf16)f.y; p[2] = (__bf16)f.z; p[3] = (__bf16)f.w;
    }
    __syncthreads();

    // B fragment: 16 contiguous bf16 at K = kb + half*16
    v16bf bfrag = *(const v16bf*)(wrow + kb + half * 16);

    for (int rs = 0; rs < 8; ++rs) {
      int r = rs * 16 + n16;
      v8bf lo = *(const v8bf*)&lA[r][half * 16];
      v8bf hi = *(const v8bf*)&lA[r][half * 16 + 8];
      v16bf afrag = __builtin_shufflevector(lo, hi, 0,1,2,3,4,5,6,7,8,9,10,11,12,13,14,15);
      acc[rs] = __builtin_amdgcn_wmma_f32_16x16x32_bf16(
          false, afrag, false, bfrag, (short)0, acc[rs], false, false);
    }
  }

  // epilogue: pure stores; uniform fast path for full tiles
  const int col = wave * 16 + n16;
  if (rowBase + 128 <= N) {
    float* hp = H + (size_t)(rowBase + half * 8) * DH + col;
    for (int rs = 0; rs < 8; ++rs) {
      for (int v = 0; v < 8; ++v) hp[(size_t)v * DH] = acc[rs][v];
      hp += (size_t)16 * DH;
    }
  } else {
    for (int rs = 0; rs < 8; ++rs)
      for (int v = 0; v < 8; ++v) {
        int row = rowBase + rs * 16 + half * 8 + v;
        if (row < N) H[(size_t)row * DH + col] = acc[rs][v];
      }
  }
}

// ---------------- edge scatter-add, layer 1 (128 dims, float4 per lane) ----------------
// gathers dinv[src]*H[src], accumulates into agg[dst]
__global__ __launch_bounds__(256) void k_agg1(const int* __restrict__ src,
                                              const int* __restrict__ dst,
                                              const float* __restrict__ H,
                                              const float* __restrict__ dinv,
                                              float* __restrict__ agg, int E) {
  int e = blockIdx.x * 8 + (threadIdx.x >> 5);
  if (e >= E) return;
  int lane = threadIdx.x & 31;
  int s = src[e], d = dst[e];
  float ds_ = dinv[s];
  float4 v = *(const float4*)(H + (size_t)s * DH + lane * 4);
  float* o = agg + (size_t)d * DH + lane * 4;
  atomicAdd(o + 0, v.x * ds_); atomicAdd(o + 1, v.y * ds_);
  atomicAdd(o + 2, v.z * ds_); atomicAdd(o + 3, v.w * ds_);
}

// h1 = relu(dinv*(agg + dinv*H) + b1), written in place over H
__global__ void k_fin1(float* __restrict__ H, const float* __restrict__ agg,
                       const float* __restrict__ dinv, const float* __restrict__ b1, int N) {
  int i = blockIdx.x * 256 + threadIdx.x;           // float4 index over N*32
  if (i >= N * 32) return;
  int row = i >> 5, c4 = (i & 31) * 4;
  float di = dinv[row];
  float4 hs = *(const float4*)(H   + (size_t)row * DH + c4);
  float4 ag = *(const float4*)(agg + (size_t)row * DH + c4);
  float4 bb = *(const float4*)(b1 + c4);
  float4 o;
  o.x = fmaxf(di * (ag.x + di * hs.x) + bb.x, 0.f);
  o.y = fmaxf(di * (ag.y + di * hs.y) + bb.y, 0.f);
  o.z = fmaxf(di * (ag.z + di * hs.z) + bb.z, 0.f);
  o.w = fmaxf(di * (ag.w + di * hs.w) + bb.w, 0.f);
  *(float4*)(H + (size_t)row * DH + c4) = o;
}

// ---------------- GEMM2: G = h1 @ W2 (unscaled), bf16 WMMA ----------------
// Wave w owns row-subtile w, computes 3 column tiles (48 padded cols).
__global__ __launch_bounds__(256) void k_gemm2(const float* __restrict__ H1,
                                               const __bf16* __restrict__ W2t,
                                               float* __restrict__ G, int N) {
  __shared__ __bf16 lA[128][48];
  const int tid  = threadIdx.x;
  const int wave = tid >> 5;
  const int lane = tid & 31;
  const int n16  = lane & 15;
  const int half = lane >> 4;
  const int rowBase = blockIdx.x * 128;

  v8f z = {};
  v8f acc[3]; acc[0] = z; acc[1] = z; acc[2] = z;

  for (int kb = 0; kb < DH; kb += 32) {
    __syncthreads();
    for (int it = 0; it < 4; ++it) {
      int i = tid + it * 256;
      int r = i >> 3, c4 = i & 7;
      int gr = rowBase + r; if (gr >= N) gr = N - 1;
      float4 f = *(const float4*)(H1 + (size_t)gr * DH + kb + c4 * 4);
      __bf16* p = &lA[r][c4 * 4];
      p[0] = (__bf16)f.x; p[1] = (__bf16)f.y; p[2] = (__bf16)f.z; p[3] = (__bf16)f.w;
    }
    __syncthreads();

    int r = wave * 16 + n16;
    v8bf lo = *(const v8bf*)&lA[r][half * 16];
    v8bf hi = *(const v8bf*)&lA[r][half * 16 + 8];
    v16bf afrag = __builtin_shufflevector(lo, hi, 0,1,2,3,4,5,6,7,8,9,10,11,12,13,14,15);

    for (int ct = 0; ct < 3; ++ct) {
      v16bf bfrag = *(const v16bf*)(W2t + (size_t)(ct * 16 + n16) * DH + kb + half * 16);
      acc[ct] = __builtin_amdgcn_wmma_f32_16x16x32_bf16(
          false, afrag, false, bfrag, (short)0, acc[ct], false, false);
    }
  }

  const int rowLane = rowBase + wave * 16 + half * 8;
  if (rowBase + 128 <= N) {
    for (int ct = 0; ct < 3; ++ct) {
      int col = ct * 16 + n16;
      if (col >= DOUT) continue;                    // lanes with col>=40 skip (pad)
      float* gp = G + (size_t)rowLane * DOUT + col;
      for (int v = 0; v < 8; ++v) gp[(size_t)v * DOUT] = acc[ct][v];
    }
  } else {
    for (int ct = 0; ct < 3; ++ct) {
      int col = ct * 16 + n16;
      if (col >= DOUT) continue;
      for (int v = 0; v < 8; ++v) {
        int row = rowLane + v;
        if (row < N) G[(size_t)row * DOUT + col] = acc[ct][v];
      }
    }
  }
}

// ---------------- edge scatter-add, layer 2 (40 dims) ----------------
__global__ __launch_bounds__(256) void k_agg2(const int* __restrict__ src,
                                              const int* __restrict__ dst,
                                              const float* __restrict__ G,
                                              const float* __restrict__ dinv,
                                              float* __restrict__ agg, int E) {
  int e = blockIdx.x * 8 + (threadIdx.x >> 5);
  if (e >= E) return;
  int lane = threadIdx.x & 31;
  int s = src[e], d = dst[e];
  float ds_ = dinv[s];
  atomicAdd(agg + (size_t)d * DOUT + lane, G[(size_t)s * DOUT + lane] * ds_);
  if (lane < 8)
    atomicAdd(agg + (size_t)d * DOUT + 32 + lane, G[(size_t)s * DOUT + 32 + lane] * ds_);
}

// h2 = dinv*(agg + dinv*G) + b2 ; out[0:N*40]=h2 ; out[N*40:]=log_softmax(h2)
__global__ __launch_bounds__(256) void k_fin2(const float* __restrict__ G,
                                              const float* __restrict__ agg,
                                              const float* __restrict__ dinv,
                                              const float* __restrict__ b2,
                                              float* __restrict__ out, int N) {
  int node = blockIdx.x * 8 + (threadIdx.x >> 5);
  if (node >= N) return;
  int lane = threadIdx.x & 31;
  bool has1 = lane < 8;
  float di = dinv[node];
  size_t base = (size_t)node * DOUT;
  float v0 = di * (agg[base + lane] + di * G[base + lane]) + b2[lane];
  float v1 = 0.f;
  if (has1) v1 = di * (agg[base + 32 + lane] + di * G[base + 32 + lane]) + b2[32 + lane];
  out[base + lane] = v0;
  if (has1) out[base + 32 + lane] = v1;

  float m = has1 ? fmaxf(v0, v1) : v0;
  for (int off = 16; off > 0; off >>= 1) m = fmaxf(m, __shfl_xor(m, off, 32));
  float s = expf(v0 - m) + (has1 ? expf(v1 - m) : 0.f);
  for (int off = 16; off > 0; off >>= 1) s += __shfl_xor(s, off, 32);
  float lse = m + logf(s);

  float* out2 = out + (size_t)N * DOUT;
  out2[base + lane] = v0 - lse;
  if (has1) out2[base + 32 + lane] = v1 - lse;
}

// ---------------- launcher ----------------
static inline size_t alignup(size_t x) { return (x + 255) & ~(size_t)255; }
static inline int cdiv(int a, int b) { return (a + b - 1) / b; }

extern "C" void kernel_launch(void* const* d_in, const int* in_sizes, int n_in,
                              void* d_out, int out_size, void* d_ws, size_t ws_size,
                              hipStream_t stream) {
  (void)n_in; (void)out_size; (void)ws_size;
  const float* X  = (const float*)d_in[0];
  const int*   EI = (const int*)  d_in[1];
  const float* W1 = (const float*)d_in[2];
  const float* b1 = (const float*)d_in[3];
  const float* W2 = (const float*)d_in[4];
  const float* b2 = (const float*)d_in[5];
  float* out = (float*)d_out;

  const int N = in_sizes[0] / DIN;
  const int E = in_sizes[1] / 2;
  const int* src = EI;
  const int* dst = EI + E;

  char* ws = (char*)d_ws;
  size_t off = 0;
  float*  deg  = (float*)(ws + off); off = alignup(off + (size_t)N * 4);
  float*  dinv = (float*)(ws + off); off = alignup(off + (size_t)N * 4);
  __bf16* W1t  = (__bf16*)(ws + off); off = alignup(off + (size_t)DH * DIN * 2);
  __bf16* W2t  = (__bf16*)(ws + off); off = alignup(off + (size_t)DOUTP * DH * 2);
  float*  H    = (float*)(ws + off); off = alignup(off + (size_t)N * DH * 4);
  float*  agg1 = (float*)(ws + off); off = alignup(off + (size_t)N * DH * 4);
  float*  G    = (float*)(ws + off); off = alignup(off + (size_t)N * DOUT * 4);
  float*  agg2 = (float*)(ws + off); off = alignup(off + (size_t)N * DOUT * 4);

  // weight conversion (bf16, transposed so B fragments are contiguous)
  k_w1t<<<cdiv(DH * DIN, 256), 256, 0, stream>>>(W1, W1t);
  k_w2t<<<cdiv(DOUTP * DH, 256), 256, 0, stream>>>(W2, W2t);

  // degree / normalization
  k_deg_init <<<cdiv(N, 256), 256, 0, stream>>>(deg, N);
  k_deg_edges<<<cdiv(E, 256), 256, 0, stream>>>(dst, deg, E);
  k_dinv     <<<cdiv(N, 256), 256, 0, stream>>>(deg, dinv, N);

  // zero aggregation buffers (every call; deterministic)
  k_fill0<<<cdiv(N * 32, 256), 256, 0, stream>>>((float4*)agg1, N * 32);
  k_fill0<<<cdiv(N * 10, 256), 256, 0, stream>>>((float4*)agg2, N * 10);

  // layer 1
  k_gemm1<<<cdiv(N, 128), 256, 0, stream>>>(X, W1t, H, N);
  k_agg1 <<<cdiv(E, 8),   256, 0, stream>>>(src, dst, H, dinv, agg1, E);
  k_fin1 <<<cdiv(N * 32, 256), 256, 0, stream>>>(H, agg1, dinv, b1, N);

  // layer 2
  k_gemm2<<<cdiv(N, 128), 256, 0, stream>>>(H, W2t, G, N);
  k_agg2 <<<cdiv(E, 8),   256, 0, stream>>>(src, dst, G, dinv, agg2, E);

  // output + log-softmax
  k_fin2<<<cdiv(N, 8), 256, 0, stream>>>(G, agg2, dinv, b2, out, N);
}
